// MoE3_4028679323874
// MI455X (gfx1250) — matrix operand order repaired and checked by reading, not compiled, not run on hardware
//
#include <hip/hip_runtime.h>
#include <hip/hip_bf16.h>

// Problem constants (from reference)
#define BB 4
#define NN 4096
#define TT (BB * NN)      // 16384 tokens
#define DD 768
#define EE 8
#define HH 3072
#define LN_EPS 1e-5f

// FFN tiling
#define TM 64             // tokens per block
#define HC 256            // H-chunk held in LDS between the two GEMMs
#define NTH 512           // threads per block (16 waves)
#define XPAD 8
#define HPAD 8

typedef __bf16 bf16;
typedef float v8f   __attribute__((ext_vector_type(8)));
typedef bf16  v4bf  __attribute__((ext_vector_type(4)));
typedef bf16  v8bf  __attribute__((ext_vector_type(8)));
typedef bf16  v16bf __attribute__((ext_vector_type(16)));

__device__ __forceinline__ v16bf cat16(v8bf lo, v8bf hi) {
    return __builtin_shufflevector(lo, hi, 0,1,2,3,4,5,6,7,8,9,10,11,12,13,14,15);
}

// B-fragment: 16 contiguous bf16 per lane (lanes 0-15: K=0..15 of column l16,
// lanes 16-31: K=16..31). Weights are stored transposed so this is contiguous.
__device__ __forceinline__ v16bf load_bfrag(const bf16* p) {
    v8bf lo = *(const v8bf*)p;
    v8bf hi = *(const v8bf*)(p + 8);
    return cat16(lo, hi);
}

// A-fragment from LDS row: lanes 0-15 hold K=[0,16) of row l16,
// lanes 16-31 hold K=[16,32)  (contiguous, matching the B-fragment).
__device__ __forceinline__ v16bf load_afrag(const bf16* rowp) {
    v8bf lo = *(const v8bf*)rowp;
    v8bf hi = *(const v8bf*)(rowp + 8);
    return cat16(lo, hi);
}

__device__ __forceinline__ float gelu_exact(float v) {
    return 0.5f * v * (1.0f + erff(v * 0.70710678118654752f));
}

__device__ __forceinline__ void pf(const void* p) { __builtin_prefetch(p, 0, 1); }

// ---------------------------------------------------------------------------
// Transpose + fp32->bf16 convert, per expert slab: in [R][C] -> out [C][R]
// R, C multiples of 32. block (32,8), grid (C/32, R/32, E)
// ---------------------------------------------------------------------------
__global__ __launch_bounds__(256) void transpose_cvt(
    const float* __restrict__ in, bf16* __restrict__ out, int R, int C)
{
    __shared__ float tile[32][33];
    const int e = blockIdx.z;
    const float* src = in + (size_t)e * R * C;
    bf16* dst = out + (size_t)e * R * C;
    const int c0 = blockIdx.x * 32, r0 = blockIdx.y * 32;
    const int tx = threadIdx.x, ty = threadIdx.y;
    #pragma unroll
    for (int i = 0; i < 32; i += 8)
        tile[ty + i][tx] = src[(size_t)(r0 + ty + i) * C + (c0 + tx)];
    __syncthreads();
    #pragma unroll
    for (int i = 0; i < 32; i += 8)
        dst[(size_t)(c0 + ty + i) * R + (r0 + tx)] = (bf16)tile[tx][ty + i];
}

// ---------------------------------------------------------------------------
// Router: one wave32 per token. Computes logits, top-2, softmax gates,
// appends (token, gate) to per-expert lists with atomics.
// grid = TT/8, block = 256 (8 waves)
// ---------------------------------------------------------------------------
__global__ __launch_bounds__(256) void router_kernel(
    const float* __restrict__ x, const float* __restrict__ rw,
    const float* __restrict__ rbias,
    int* __restrict__ counts, int* __restrict__ tok_ids, float* __restrict__ gates)
{
    const int lane = threadIdx.x & 31;
    const int t = blockIdx.x * 8 + (threadIdx.x >> 5);
    const float* xr = x + (size_t)t * DD;

    float acc[EE];
    #pragma unroll
    for (int e = 0; e < EE; ++e) acc[e] = 0.0f;

    #pragma unroll 4
    for (int j = 0; j < DD / 32; ++j) {
        const int d = j * 32 + lane;
        const float xv = xr[d];
        const float4* w4 = (const float4*)(rw + (size_t)d * EE);
        float4 a = w4[0], b = w4[1];
        acc[0] += xv * a.x; acc[1] += xv * a.y; acc[2] += xv * a.z; acc[3] += xv * a.w;
        acc[4] += xv * b.x; acc[5] += xv * b.y; acc[6] += xv * b.z; acc[7] += xv * b.w;
    }
    #pragma unroll
    for (int off = 16; off >= 1; off >>= 1)
        #pragma unroll
        for (int e = 0; e < EE; ++e)
            acc[e] += __shfl_xor(acc[e], off, 32);

    if (lane == 0) {
        float lg[EE];
        #pragma unroll
        for (int e = 0; e < EE; ++e) lg[e] = acc[e] + rbias[e];
        int e0 = 0;
        #pragma unroll
        for (int e = 1; e < EE; ++e) if (lg[e] > lg[e0]) e0 = e;
        int e1 = (e0 == 0) ? 1 : 0;
        #pragma unroll
        for (int e = 0; e < EE; ++e) if (e != e1 && e != e0 && lg[e] > lg[e1]) e1 = e;
        const float ex = expf(lg[e1] - lg[e0]);   // <= 1
        const float w0 = 1.0f / (1.0f + ex);
        const float w1 = ex * w0;
        int p0 = atomicAdd(&counts[e0], 1);
        tok_ids[e0 * TT + p0] = t;  gates[e0 * TT + p0] = w0;
        int p1 = atomicAdd(&counts[e1], 1);
        tok_ids[e1 * TT + p1] = t;  gates[e1 * TT + p1] = w1;
    }
}

// ---------------------------------------------------------------------------
// Fused expert FFN: out += gate * ( gelu(X @ W1 + b1) @ W2 + b2 )
// 64 gathered tokens per block, 16 waves, bf16 WMMA with fp32 accumulation.
// w1t: [E][H][D] (transposed), w2t: [E][D][H] (transposed).
// grid (TT/64, E), block 512.
//
// Phase A per H-chunk: wave w computes Ht[:, w*16..+16) over all 64 rows --
//   one B-fragment is reused by 4 row-block WMMAs.
// Phase B: wave w owns output cols [w*48, w*48+48): 3 col-tiles x 4 row-blocks
//   = 12 persistent f32 accumulators (96 VGPRs).
// ---------------------------------------------------------------------------
__global__ __launch_bounds__(NTH, 1) void ffn_kernel(
    const float* __restrict__ x,
    const bf16* __restrict__ w1t, const bf16* __restrict__ w2t,
    const float* __restrict__ b1, const float* __restrict__ b2,
    const int* __restrict__ counts, const int* __restrict__ tok_ids,
    const float* __restrict__ gates, float* __restrict__ out)
{
    const int e = blockIdx.y;
    const int cnt = counts[e];
    const int row0 = blockIdx.x * TM;
    if (row0 >= cnt) return;

    __shared__ bf16 Xs[TM][DD + XPAD];   // 99328 B
    __shared__ bf16 Hs[TM][HC + HPAD];   // 33792 B
    __shared__ int   stok[TM];
    __shared__ float sgate[TM];

    const int tid  = threadIdx.x;
    const int wave = tid >> 5;           // 0..15
    const int lane = tid & 31;
    const int lhalf = lane >> 4;         // 0 | 1
    const int l16   = lane & 15;

    if (tid < TM) {
        const int idx = row0 + tid;
        if (idx < cnt) { stok[tid] = tok_ids[e * TT + idx]; sgate[tid] = gates[e * TT + idx]; }
        else           { stok[tid] = -1;                    sgate[tid] = 0.0f; }
    }
    __syncthreads();

    // Gather 64 token rows (fp32 -> bf16, float4-vectorized) into LDS
    for (int i = tid * 4; i < TM * DD; i += NTH * 4) {
        const int r = i / DD, c = i % DD;
        const int t = stok[r];
        v4bf v;
        if (t >= 0) {
            const float4 xv = *(const float4*)(x + (size_t)t * DD + c);
            v[0] = (bf16)xv.x; v[1] = (bf16)xv.y; v[2] = (bf16)xv.z; v[3] = (bf16)xv.w;
        } else {
            v[0] = v[1] = v[2] = v[3] = (bf16)0.0f;
        }
        *(v4bf*)&Xs[r][c] = v;
    }
    __syncthreads();

    const bf16* w1e = w1t + (size_t)e * HH * DD;
    const bf16* w2e = w2t + (size_t)e * HH * DD;
    const int n0 = wave * 48;            // phase-B output column base for this wave

    v8f yacc[4][3];
    #pragma unroll
    for (int rb = 0; rb < 4; ++rb)
        #pragma unroll
        for (int t = 0; t < 3; ++t) yacc[rb][t] = (v8f){};

    for (int hc = 0; hc < HH; hc += HC) {
        // Prefetch next chunk's weight lines into cache (global_prefetch_b8)
        if (hc + HC < HH) {
            const bf16* nb1 = w1e + (size_t)(hc + HC + wave * 16 + l16) * DD;
            pf(nb1); pf(nb1 + 256); pf(nb1 + 512);
            const bf16* nb2 = w2e + (size_t)(n0 + l16) * HH + hc + HC;
            pf(nb2); pf(nb2 + (size_t)16 * HH); pf(nb2 + (size_t)32 * HH);
        }

        // ---- Phase A: Ht[:, wave*16 .. +16) = X @ W1 chunk (all 64 rows) ----
        const int hcol0 = hc + wave * 16;
        const bf16* bbase = w1e + (size_t)(hcol0 + l16) * DD + lhalf * 16;
        v8f ha[4];
        #pragma unroll
        for (int rb = 0; rb < 4; ++rb) ha[rb] = (v8f){};

        v16bf bcur = load_bfrag(bbase);                  // software-pipelined B
        #pragma unroll 2
        for (int k = 0; k < DD; k += 32) {
            const int knext = (k + 32 < DD) ? (k + 32) : 0;
            const v16bf bnxt = load_bfrag(bbase + knext);
            v16bf af[4];
            #pragma unroll
            for (int rb = 0; rb < 4; ++rb)
                af[rb] = load_afrag(&Xs[rb * 16 + l16][k + lhalf * 16]);
            #pragma unroll
            for (int rb = 0; rb < 4; ++rb)
                ha[rb] = __builtin_amdgcn_wmma_f32_16x16x32_bf16(
                    false, af[rb], false, bcur, (short)0, ha[rb], false, false);
            bcur = bnxt;
        }
        // bias + exact GELU + store bf16 to LDS
        const float bbias = b1[e * HH + hcol0 + l16];
        #pragma unroll
        for (int rb = 0; rb < 4; ++rb)
            #pragma unroll
            for (int r = 0; r < 8; ++r) {
                const float v = gelu_exact(ha[rb][r] + bbias);
                Hs[rb * 16 + r + lhalf * 8][wave * 16 + l16] = (bf16)v;
            }
        __syncthreads();

        // ---- Phase B: Y[:, n0 .. n0+48) += gelu(Ht) @ W2 chunk ----
        #pragma unroll
        for (int k2 = 0; k2 < HC; k2 += 32) {
            v16bf bfr[3];
            #pragma unroll
            for (int t = 0; t < 3; ++t)
                bfr[t] = load_bfrag(w2e + (size_t)(n0 + t * 16 + l16) * HH
                                    + hc + k2 + lhalf * 16);
            v16bf af[4];
            #pragma unroll
            for (int rb = 0; rb < 4; ++rb)
                af[rb] = load_afrag(&Hs[rb * 16 + l16][k2 + lhalf * 16]);
            #pragma unroll
            for (int t = 0; t < 3; ++t)
                #pragma unroll
                for (int rb = 0; rb < 4; ++rb)
                    yacc[rb][t] = __builtin_amdgcn_wmma_f32_16x16x32_bf16(
                        false, af[rb], false, bfr[t], (short)0, yacc[rb][t], false, false);
        }
        __syncthreads();
    }

    // ---- Epilogue: + b2, * gate, atomic add into output accumulator ----
    #pragma unroll
    for (int rb = 0; rb < 4; ++rb)
        #pragma unroll
        for (int r = 0; r < 8; ++r) {
            const int m = rb * 16 + r + lhalf * 8;
            const int t = stok[m];
            if (t < 0) continue;
            const float g = sgate[m];
            float* orow = out + (size_t)t * DD;
            #pragma unroll
            for (int tt = 0; tt < 3; ++tt) {
                const int n = n0 + tt * 16 + l16;
                atomicAdd(&orow[n], (yacc[rb][tt][r] + b2[e * DD + n]) * g);
            }
        }
}

// ---------------------------------------------------------------------------
// Residual + LayerNorm. One block (256 threads) per token row; `out` holds the
// MoE accumulator on entry and the final result on exit.
// ---------------------------------------------------------------------------
__global__ __launch_bounds__(256) void ln_kernel(
    const float* __restrict__ x, const float* __restrict__ gamma,
    const float* __restrict__ beta, float* __restrict__ out)
{
    const int t = blockIdx.x;
    const float* xr = x + (size_t)t * DD;
    float* orow = out + (size_t)t * DD;
    const int tid = threadIdx.x, wave = tid >> 5, lane = tid & 31;

    float v[3], s = 0.0f, q = 0.0f;
    #pragma unroll
    for (int i = 0; i < 3; ++i) {
        const int d = tid + 256 * i;
        const float y = xr[d] + orow[d];
        v[i] = y; s += y; q += y * y;
    }
    #pragma unroll
    for (int off = 16; off >= 1; off >>= 1) {
        s += __shfl_xor(s, off, 32);
        q += __shfl_xor(q, off, 32);
    }
    __shared__ float sS[8], sQ[8];
    if (lane == 0) { sS[wave] = s; sQ[wave] = q; }
    __syncthreads();
    if (tid == 0) {
        float S = 0.0f, Q = 0.0f;
        #pragma unroll
        for (int i = 0; i < 8; ++i) { S += sS[i]; Q += sQ[i]; }
        sS[0] = S; sQ[0] = Q;
    }
    __syncthreads();
    const float mu = sS[0] * (1.0f / DD);
    const float var = sQ[0] * (1.0f / DD) - mu * mu;
    const float rstd = rsqrtf(var + LN_EPS);
    #pragma unroll
    for (int i = 0; i < 3; ++i) {
        const int d = tid + 256 * i;
        orow[d] = (v[i] - mu) * rstd * gamma[d] + beta[d];
    }
}

// ---------------------------------------------------------------------------
extern "C" void kernel_launch(void* const* d_in, const int* in_sizes, int n_in,
                              void* d_out, int out_size, void* d_ws, size_t ws_size,
                              hipStream_t stream)
{
    const float* x        = (const float*)d_in[0];
    const float* router_w = (const float*)d_in[1];
    const float* router_b = (const float*)d_in[2];
    const float* w1       = (const float*)d_in[3];
    const float* b1       = (const float*)d_in[4];
    const float* w2       = (const float*)d_in[5];
    const float* b2       = (const float*)d_in[6];
    const float* gamma    = (const float*)d_in[7];
    const float* beta     = (const float*)d_in[8];
    float* out            = (float*)d_out;

    char* ws = (char*)d_ws;
    size_t o = 0;
    bf16* w1t = (bf16*)(ws + o); o += (size_t)EE * HH * DD * sizeof(bf16);
    bf16* w2t = (bf16*)(ws + o); o += (size_t)EE * HH * DD * sizeof(bf16);
    int*   counts  = (int*)(ws + o);   o += 256;
    int*   tok_ids = (int*)(ws + o);   o += (size_t)EE * TT * sizeof(int);
    float* gatesb  = (float*)(ws + o); o += (size_t)EE * TT * sizeof(float);

    // Zero output accumulator and expert counts (deterministic each call)
    hipMemsetAsync(d_out, 0, (size_t)TT * DD * sizeof(float), stream);
    hipMemsetAsync(counts, 0, 256, stream);

    // Weight convert+transpose: w1 [E][D][H] -> w1t [E][H][D]; w2 [E][H][D] -> w2t [E][D][H]
    transpose_cvt<<<dim3(HH / 32, DD / 32, EE), dim3(32, 8), 0, stream>>>(w1, w1t, DD, HH);
    transpose_cvt<<<dim3(DD / 32, HH / 32, EE), dim3(32, 8), 0, stream>>>(w2, w2t, HH, DD);

    // Router + top-2 gating + expert token lists
    router_kernel<<<TT / 8, 256, 0, stream>>>(x, router_w, router_b,
                                              counts, tok_ids, gatesb);

    // Fused expert FFN over gathered token tiles (bf16 WMMA)
    ffn_kernel<<<dim3(TT / TM, EE), NTH, 0, stream>>>(x, w1t, w2t, b1, b2,
                                                      counts, tok_ids, gatesb, out);

    // Residual + LayerNorm
    ln_kernel<<<TT, 256, 0, stream>>>(x, gamma, beta, out);
}
